// SemiLocalAttention_4587025072276
// MI455X (gfx1250) — compile-verified, matrix operand order checked by reading, not run on hardware
//
#include <hip/hip_runtime.h>

// ---------------------------------------------------------------------------
// SemiLocalAttention for MI455X (gfx1250, wave32, WMMA).
//
// Pipeline:
//   1) qkv = x @ w_qkv^T + b_qkv          (WMMA f16 GEMM, fp32 accum)
//   2) u   = x @ w_u^T   + b_u            (WMMA f16 GEMM)
//   3) g   = silu(semi_local_attn(q,k,v)) * u   (flash-style masked WMMA attn)
//   4) out = g @ w_out^T + b_out          (WMMA f16 GEMM)
//
// Mask structure (K1=K2=64): per query i, keys are the union of
//   global prefix  j <= min(K2, L-i, i)
//   anti-diag band max(0, L-K1-i) <= j <= min(i, L-i)
// so each 16-row query tile touches <= ~12 key chunks of 32 -> ~1/16 of
// dense attention FLOPs. Total ~26 GFLOP vs ~200 MB traffic: near the
// 23.3 TB/s memory roofline, so f16 WMMA w/ fp32 accum + single-pass
// streaming of fp32 tensors is the right operating point.
// ---------------------------------------------------------------------------

#define B_  4
#define L_  2048
#define D_  512
#define H_  8
#define HD_ 64
#define K1_ 64
#define K2_ 64

typedef __attribute__((ext_vector_type(16))) _Float16 v16h;
typedef __attribute__((ext_vector_type(8)))  _Float16 v8h;
typedef __attribute__((ext_vector_type(8)))  float    v8f;
typedef __attribute__((ext_vector_type(4)))  float    f4;

__device__ __forceinline__ float expx(float x) {
  // exp(x) = exp2(x * log2(e)); lowers to v_exp_f32
  return __builtin_amdgcn_exp2f(x * 1.4426950408889634f);
}

__device__ __forceinline__ v8f wmma16(v16h a, v16h b, v8f c) {
  // D = A(16x32 f16) * B(32x16 f16) + C(16x16 f32)
  return __builtin_amdgcn_wmma_f32_16x16x32_f16(false, a, false, b, (short)0, c,
                                                false, false);
}

// A-fragment (16x32 f16) load from a row-major fp32 matrix.
// ISA 7.12.2 wave32 layout: lane = half*16 + m (m = row 0..15, half = k-half),
// VGPR v(0..3) <- k = half*8 + 2v..2v+1 ; VGPR v(4..7) <- k = 16 + half*8 + ...
// => vector element e:  e in [0,8)  -> col half*8 + e
//                       e in [8,16) -> col 16 + half*8 + (e-8)
__device__ __forceinline__ v16h frag_a_g(const float* __restrict__ base,
                                         int pitch, int lane, float scl) {
  const int m = lane & 15, half = lane >> 4;
  const float* r = base + (size_t)m * pitch + half * 8;
  f4 a = *(const f4*)(r);
  f4 b = *(const f4*)(r + 4);
  f4 c = *(const f4*)(r + 16);
  f4 d = *(const f4*)(r + 20);
  v16h f;
#pragma unroll
  for (int t = 0; t < 4; ++t) {
    f[t]      = (_Float16)(a[t] * scl);
    f[t + 4]  = (_Float16)(b[t] * scl);
    f[t + 8]  = (_Float16)(c[t] * scl);
    f[t + 12] = (_Float16)(d[t] * scl);
  }
  return f;
}

// Same fragment pattern from a row-major f16 matrix in LDS.
__device__ __forceinline__ v16h frag_a_lds(const _Float16* base, int pitch,
                                           int lane) {
  const int m = lane & 15, half = lane >> 4;
  const _Float16* r = base + m * pitch + half * 8;
  v8h lo = *(const v8h*)(r);
  v8h hi = *(const v8h*)(r + 16);
  v16h f;
#pragma unroll
  for (int t = 0; t < 8; ++t) { f[t] = lo[t]; f[t + 8] = hi[t]; }
  return f;
}

// ---------------------------------------------------------------------------
// GEMM: C[M,N] = A[M,K] @ W[N,K]^T + bias[N]   (fp32 in/out, f16 WMMA compute)
// N, K are template constants -> row strides fold into immediate offsets.
// Block: 256 threads = 8 waves in 2x4; tile 64x128; K-step 32.
// Register prefetch + LDS ping-pong: one barrier per K-step; safe because
// __syncthreads() lowers to s_wait_dscnt 0 + s_barrier, so all reads of a
// buffer complete before any writer passes the next barrier.
// B-matrix = W^T, so B^T = W: B-fragments load A-style from row-major W tile.
// Requires M%64==0, N%128==0, K%32==0 (true for all three uses here).
// ---------------------------------------------------------------------------
template <int N, int K>
__global__ __launch_bounds__(256)
void gemm_xwt_bias_kernel(const float* __restrict__ A, const float* __restrict__ W,
                          const float* __restrict__ bias, float* __restrict__ C,
                          int M) {
  __shared__ _Float16 As[2][64 * 32];    // 2 x 4 KB
  __shared__ _Float16 Ws[2][128 * 32];   // 2 x 8 KB
  const int tid  = threadIdx.x;
  const int lane = tid & 31;
  const int wvid = tid >> 5;
  const int wm   = wvid >> 2;            // 0..1 (32-row slab)
  const int wn   = wvid & 3;             // 0..3 (32-col slab)
  const int m0   = blockIdx.x * 64;
  const int n0   = blockIdx.y * 128;

  const int arow = tid >> 2, acol = (tid & 3) * 8;   // A tile: 8 f32/thread
  const int wrow = tid >> 1, wcol = (tid & 1) * 16;  // W tile: 16 f32/thread
  const float* aptr = A + (size_t)(m0 + arow) * K + acol;
  const float* wptr = W + (size_t)(n0 + wrow) * K + wcol;

  // Preload K-tile 0 into registers.
  f4 ra0 = *(const f4*)(aptr);
  f4 ra1 = *(const f4*)(aptr + 4);
  f4 rw0 = *(const f4*)(wptr);
  f4 rw1 = *(const f4*)(wptr + 4);
  f4 rw2 = *(const f4*)(wptr + 8);
  f4 rw3 = *(const f4*)(wptr + 12);

  v8f c00 = {}, c01 = {}, c10 = {}, c11 = {};

  int p = 0;
#pragma unroll 2
  for (int k0 = 0; k0 < K; k0 += 32, p ^= 1) {
    // Convert + park current K-tile into LDS[p].
    {
      v8h ha, hw0, hw1;
#pragma unroll
      for (int t = 0; t < 4; ++t) {
        ha[t]  = (_Float16)ra0[t]; ha[t + 4]  = (_Float16)ra1[t];
        hw0[t] = (_Float16)rw0[t]; hw0[t + 4] = (_Float16)rw1[t];
        hw1[t] = (_Float16)rw2[t]; hw1[t + 4] = (_Float16)rw3[t];
      }
      *(v8h*)(&As[p][arow * 32 + acol])     = ha;
      *(v8h*)(&Ws[p][wrow * 32 + wcol])     = hw0;
      *(v8h*)(&Ws[p][wrow * 32 + wcol + 8]) = hw1;
    }
    __syncthreads();

    // Issue global loads for the next K-tile (overlaps with WMMA below),
    // and prefetch the tile after that into L2 (global_prefetch_b8).
    if (k0 + 32 < K) {
      const float* an = aptr + k0 + 32;
      const float* wq = wptr + k0 + 32;
      ra0 = *(const f4*)(an);     ra1 = *(const f4*)(an + 4);
      rw0 = *(const f4*)(wq);     rw1 = *(const f4*)(wq + 4);
      rw2 = *(const f4*)(wq + 8); rw3 = *(const f4*)(wq + 12);
      if (k0 + 64 < K) {
        __builtin_prefetch(aptr + k0 + 64, 0, 1);
        __builtin_prefetch(wptr + k0 + 64, 0, 1);
      }
    }

    v16h a0 = frag_a_lds(&As[p][(wm * 32 +  0) * 32], 32, lane);
    v16h a1 = frag_a_lds(&As[p][(wm * 32 + 16) * 32], 32, lane);
    v16h b0 = frag_a_lds(&Ws[p][(wn * 32 +  0) * 32], 32, lane);
    v16h b1 = frag_a_lds(&Ws[p][(wn * 32 + 16) * 32], 32, lane);
    c00 = wmma16(a0, b0, c00);
    c01 = wmma16(a0, b1, c01);
    c10 = wmma16(a1, b0, c10);
    c11 = wmma16(a1, b1, c11);
  }

  // C/D layout (ISA 7.12.2): lane holds col n = lane&15, rows (lane>>4)*8 + r.
  const int col = lane & 15, rb = (lane >> 4) * 8;
#pragma unroll
  for (int i = 0; i < 2; ++i) {
#pragma unroll
    for (int j = 0; j < 2; ++j) {
      v8f c = (i == 0) ? (j == 0 ? c00 : c01) : (j == 0 ? c10 : c11);
      const int n  = n0 + wn * 32 + j * 16 + col;
      const float bn = bias[n];
      const int mr = m0 + wm * 32 + i * 16 + rb;
      float* cp = C + (size_t)mr * N + n;   // row stride N is constexpr
#pragma unroll
      for (int r = 0; r < 8; ++r)
        cp[r * N] = c[r] + bn;
    }
  }
}

// ---------------------------------------------------------------------------
// Semi-local masked attention, fused with silu-gate.
// One wave per (b, h, 16-row q-tile); 4 waves per block; all lanes active
// everywhere (WMMA requires EXEC all ones). No cross-wave sharing.
// ---------------------------------------------------------------------------
__device__ __forceinline__ bool maskv(int i, int j) {
  const bool causal = j <= i;
  const bool band   = (i + j >= L_ - K1_) && (i + j <= L_);
  const bool glob   = (j <= K2_) && (j <= L_ - i);
  return causal && (band || glob);
}

__global__ __launch_bounds__(128)
void semi_local_attn_kernel(const float* __restrict__ qkv,
                            const float* __restrict__ u,
                            float* __restrict__ g) {
  __shared__ _Float16 Pbuf[4][16 * 32];  // per-wave P round-trip (1 KB each)
  __shared__ _Float16 Vt[4][64 * 32];    // per-wave transposed V chunk (4 KB)
  const int lane = threadIdx.x & 31;
  const int wv   = threadIdx.x >> 5;
  const int tile = blockIdx.x * 4 + wv;          // 0..4095
  const int qt   = tile & 127;
  const int h    = (tile >> 7) & 7;
  const int b    = tile >> 10;
  const int i0   = qt * 16;

  const int pitch = 3 * D_;  // floats per sequence position in qkv buffer
  const float* qb = qkv + (size_t)(b * L_ + i0) * pitch + 0 * D_ + h * HD_;
  const float* kb = qkv + (size_t)(b * L_) * pitch + 1 * D_ + h * HD_;
  const float* vb = qkv + (size_t)(b * L_) * pitch + 2 * D_ + h * HD_;

  const float scale = 0.125f;  // 1/sqrt(HD)
  const v16h aq0 = frag_a_g(qb,      pitch, lane, scale);  // d 0..31
  const v16h aq1 = frag_a_g(qb + 32, pitch, lane, scale);  // d 32..63

  float mrun[8], srun[8];
  v8f o[4] = {};
#pragma unroll
  for (int r = 0; r < 8; ++r) { mrun[r] = -1.0e30f; srun[r] = 0.0f; }

  const int col = lane & 15;
  const int rb  = (lane >> 4) * 8;
  _Float16* pb = Pbuf[wv];
  _Float16* vt = Vt[wv];

  auto process = [&](int j0) {
    // ---- S = Q K^T for 32 keys (two 16-col fragments) -------------------
    const float* kr = kb + (size_t)j0 * pitch;
    v16h bk00 = frag_a_g(kr,                           pitch, lane, 1.0f);
    v16h bk01 = frag_a_g(kr + 32,                      pitch, lane, 1.0f);
    v16h bk10 = frag_a_g(kr + (size_t)16 * pitch,      pitch, lane, 1.0f);
    v16h bk11 = frag_a_g(kr + (size_t)16 * pitch + 32, pitch, lane, 1.0f);
    v8f s0 = {}, s1 = {};
    s0 = wmma16(aq0, bk00, s0);
    s0 = wmma16(aq1, bk01, s0);
    s1 = wmma16(aq0, bk10, s1);
    s1 = wmma16(aq1, bk11, s1);

    // ---- stage V chunk transposed: Vt[d][jj] f16 ------------------------
    const float* vr = vb + (size_t)(j0 + lane) * pitch;
#pragma unroll
    for (int d = 0; d < 64; d += 4) {
      f4 vvv = *(const f4*)(vr + d);
      vt[(d + 0) * 32 + lane] = (_Float16)vvv[0];
      vt[(d + 1) * 32 + lane] = (_Float16)vvv[1];
      vt[(d + 2) * 32 + lane] = (_Float16)vvv[2];
      vt[(d + 3) * 32 + lane] = (_Float16)vvv[3];
    }

    // ---- exact mask + streaming softmax ---------------------------------
    // (row stats replicated across each 16-lane group; chunk j0=0 always
    //  processed first and every row has valid key j=0 => no NaNs)
#pragma unroll
    for (int r = 0; r < 8; ++r) {
      const int i = i0 + rb + r;
      float x0 = maskv(i, j0 + col)      ? s0[r] : -1.0e30f;
      float x1 = maskv(i, j0 + 16 + col) ? s1[r] : -1.0e30f;
      float mx = fmaxf(x0, x1);
#pragma unroll
      for (int off = 8; off > 0; off >>= 1) mx = fmaxf(mx, __shfl_xor(mx, off, 16));
      const float mnew = fmaxf(mrun[r], mx);
      const float corr = expx(mrun[r] - mnew);
      const float p0 = expx(x0 - mnew);
      const float p1 = expx(x1 - mnew);
      float ps = p0 + p1;
#pragma unroll
      for (int off = 8; off > 0; off >>= 1) ps += __shfl_xor(ps, off, 16);
      srun[r] = srun[r] * corr + ps;
      mrun[r] = mnew;
      o[0][r] *= corr; o[1][r] *= corr; o[2][r] *= corr; o[3][r] *= corr;
      pb[(rb + r) * 32 + col]      = (_Float16)p0;   // C-layout -> row-major
      pb[(rb + r) * 32 + 16 + col] = (_Float16)p1;
    }

    // ---- O += P V  (P via LDS round-trip, V^T fragments A-style) --------
    const v16h ap = frag_a_lds(pb, 32, lane);
#pragma unroll
    for (int t = 0; t < 4; ++t) {
      const v16h bvf = frag_a_lds(&vt[t * 16 * 32], 32, lane);
      o[t] = wmma16(ap, bvf, o[t]);
    }
  };

  // Chunk enumeration (uniform across the wave).
  const int imax = i0 + 15;
  int gend = min(min(K2_, imax), L_ - i0);   // union of per-row global maxima
  const int glast = gend & ~31;
  for (int j0 = 0; j0 <= glast; j0 += 32) process(j0);
  const int bl = max(0, L_ - K1_ - imax);
  const int bu = min(imax, L_ - i0);
  if (bl <= bu) {
    int bstart = bl & ~31;
    if (bstart <= glast) bstart = glast + 32;  // dedupe vs global chunks
    for (int j0 = bstart; j0 <= bu; j0 += 32) process(j0);
  }

  // ---- epilogue: normalize, silu-gate with u, store g[B,L,D] ------------
  const float* ub = u + (size_t)(b * L_) * D_ + h * HD_;
  float* gb = g + (size_t)(b * L_) * D_ + h * HD_;
#pragma unroll
  for (int t = 0; t < 4; ++t) {
    const int dc = t * 16 + col;
#pragma unroll
    for (int r = 0; r < 8; ++r) {
      const int i = i0 + rb + r;
      const float ov  = o[t][r] / srun[r];
      const float sig = 1.0f / (1.0f + expx(-ov));
      gb[(size_t)i * D_ + dc] = ov * sig * ub[(size_t)i * D_ + dc];
    }
  }
}

// ---------------------------------------------------------------------------
extern "C" void kernel_launch(void* const* d_in, const int* in_sizes, int n_in,
                              void* d_out, int out_size, void* d_ws, size_t ws_size,
                              hipStream_t stream) {
  (void)in_sizes; (void)n_in; (void)out_size; (void)ws_size;
  const float* x     = (const float*)d_in[0];
  const float* w_qkv = (const float*)d_in[1];
  const float* b_qkv = (const float*)d_in[2];
  const float* w_u   = (const float*)d_in[3];
  const float* b_u   = (const float*)d_in[4];
  const float* w_out = (const float*)d_in[5];
  const float* b_out = (const float*)d_in[6];
  float* out = (float*)d_out;

  const int M = B_ * L_;                             // 8192 rows
  float* qkvbuf = (float*)d_ws;                      // [B,L,3*D]  48 MB
  float* ubuf   = qkvbuf + (size_t)M * 3 * D_;       // [B,L,D]    16 MB
  float* gbuf   = ubuf   + (size_t)M * D_;           // [B,L,D]    16 MB

  gemm_xwt_bias_kernel<3 * D_, D_><<<dim3(M / 64, (3 * D_) / 128), dim3(256), 0, stream>>>(
      x, w_qkv, b_qkv, qkvbuf, M);
  gemm_xwt_bias_kernel<D_, D_><<<dim3(M / 64, D_ / 128), dim3(256), 0, stream>>>(
      x, w_u, b_u, ubuf, M);
  semi_local_attn_kernel<<<dim3(B_ * H_ * (L_ / 16) / 4), dim3(128), 0, stream>>>(
      qkvbuf, ubuf, gbuf);
  gemm_xwt_bias_kernel<D_, D_><<<dim3(M / 64, D_ / 128), dim3(256), 0, stream>>>(
      gbuf, w_out, b_out, out, M);
}
